// ColorGNNBigger_59287728554043
// MI455X (gfx1250) — matrix operand order repaired and checked by reading, not compiled, not run on hardware
//
#include <hip/hip_runtime.h>
#include <hip/hip_bf16.h>

#define NN 50000
#define NE 1600000

typedef __attribute__((ext_vector_type(8)))  unsigned short v8u;
typedef __attribute__((ext_vector_type(16))) unsigned short v16u;
typedef __attribute__((ext_vector_type(16))) __bf16        v16bf;
typedef __attribute__((ext_vector_type(8)))  float         v8f;

__device__ __forceinline__ unsigned short f2bf(float f) {
  unsigned u = __float_as_uint(f);
  unsigned r = u + 0x7FFFu + ((u >> 16) & 1u);
  return (unsigned short)(r >> 16);
}

// ---------------- precompute kernels ----------------

__global__ __launch_bounds__(256) void cvt_f32_to_bf16(const float* __restrict__ in,
                                                       unsigned short* __restrict__ out,
                                                       size_t total) {
  for (size_t i = (size_t)blockIdx.x * blockDim.x + threadIdx.x; i < total;
       i += (size_t)gridDim.x * blockDim.x)
    out[i] = f2bf(in[i]);
}

// W[K][N] (row-major, f32) -> Wt[N][K] (row-major, bf16 bits)
__global__ __launch_bounds__(256) void wt_transpose(const float* __restrict__ W,
                                                    unsigned short* __restrict__ Wt,
                                                    int K, int N) {
  size_t total = (size_t)K * N;
  for (size_t i = (size_t)blockIdx.x * blockDim.x + threadIdx.x; i < total;
       i += (size_t)gridDim.x * blockDim.x) {
    int n = (int)(i / K);
    int k = (int)(i % K);
    Wt[i] = f2bf(W[(size_t)k * N + n]);
  }
}

__global__ __launch_bounds__(256) void fill_ones(float* __restrict__ p, int n) {
  int i = blockIdx.x * blockDim.x + threadIdx.x;
  if (i < n) p[i] = 1.0f;  // self-loop weight
}

__global__ __launch_bounds__(256) void deg_scatter(float* __restrict__ deg,
                                                   const int* __restrict__ dst,
                                                   const float* __restrict__ ew, int e) {
  int i = blockIdx.x * blockDim.x + threadIdx.x;
  if (i < e) atomicAdd(&deg[dst[i]], ew[i]);
}

__global__ __launch_bounds__(256) void dinv_kernel(const float* __restrict__ deg,
                                                   float* __restrict__ dinv,
                                                   float* __restrict__ dinv2, int n) {
  int i = blockIdx.x * blockDim.x + threadIdx.x;
  if (i >= n) return;
  float d = deg[i];
  float r = (d > 0.0f) ? rsqrtf(d) : 0.0f;
  dinv[i] = r;
  dinv2[i] = r * r;  // self-loop norm: dinv * 1 * dinv
}

__global__ __launch_bounds__(256) void norm_kernel(const float* __restrict__ dinv,
                                                   const int* __restrict__ src,
                                                   const int* __restrict__ dst,
                                                   const float* __restrict__ ew,
                                                   float* __restrict__ nrm, int e) {
  int i = blockIdx.x * blockDim.x + threadIdx.x;
  if (i < e) nrm[i] = dinv[src[i]] * ew[i] * dinv[dst[i]];
}

// ---------------- WMMA GEMM: C[M,N] f32 = A[M,K] bf16 * Bt[N,K]^T bf16 ---------------
// 256 threads = 8 waves. Wave w computes a 16-row strip x 64 cols (4 accumulators).
// Grid: x = N/64, y = ceil(M/128).

__global__ __launch_bounds__(256) void gemm_bf16_wmma(const unsigned short* __restrict__ A,
                                                      const unsigned short* __restrict__ Bt,
                                                      float* __restrict__ C,
                                                      int M, int N, int K) {
  const int wave = threadIdx.x >> 5;
  const int lane = threadIdx.x & 31;
  const int g    = lane >> 4;   // half-wave select
  const int l16  = lane & 15;

  const int strip = blockIdx.y * 8 + wave;   // 16-row strip
  if (strip * 16 >= M) return;               // wave-uniform exit (EXEC stays all-ones)
  const int row     = strip * 16 + l16;
  const int colbase = blockIdx.x * 64;

  v8f acc[4] = {};
  const unsigned short* arow = A + (size_t)row * K;

  for (int k0 = 0; k0 < K; k0 += 32) {
    // A fragment, 16x32 bf16: lanes 0-15 K {0-7,16-23}, lanes 16-31 K {8-15,24-31}
    v8u alo = *(const v8u*)(arow + k0 + g * 8);
    v8u ahi = *(const v8u*)(arow + k0 + 16 + g * 8);
    if (k0 + 32 < K) __builtin_prefetch(arow + k0 + 32 + g * 8, 0, 3);
    v16u au;
#pragma unroll
    for (int i = 0; i < 8; ++i) { au[i] = alo[i]; au[8 + i] = ahi[i]; }
    v16bf af = __builtin_bit_cast(v16bf, au);

#pragma unroll
    for (int t = 0; t < 4; ++t) {
      // B fragment, 32x16 bf16 from transposed weights: lane holds col=l16,
      // K {g*8 + 0-7, 16 + g*8 + 0-7} in Wt[col][:]
      const unsigned short* bcol =
          Bt + (size_t)(colbase + t * 16 + l16) * K + k0 + g * 8;
      v8u blo = *(const v8u*)(bcol);
      v8u bhi = *(const v8u*)(bcol + 16);
      v16u bu;
#pragma unroll
      for (int i = 0; i < 8; ++i) { bu[i] = blo[i]; bu[8 + i] = bhi[i]; }
      v16bf bfv = __builtin_bit_cast(v16bf, bu);

      acc[t] = __builtin_amdgcn_wmma_f32_16x16x32_bf16(
          /*neg_a=*/false, af, /*neg_b=*/false, bfv,
          /*c_mod=*/(short)0, acc[t], /*reuse_a=*/false, /*reuse_b=*/false);
    }
  }

  // C/D layout: VGPR r, lane L: row = strip*16 + r + 8*g, col = colbase + t*16 + l16
#pragma unroll
  for (int t = 0; t < 4; ++t) {
#pragma unroll
    for (int r = 0; r < 8; ++r) {
      int m = strip * 16 + r + 8 * g;
      C[(size_t)m * N + colbase + t * 16 + l16] = acc[t][r];
    }
  }
}

// ---------------- aggregation kernels ----------------

// agg[n,f] = h[n,f]*dinv2[n] + b[f]   (self-loop contribution + bias)
__global__ __launch_bounds__(256) void init_agg(const float* __restrict__ h,
                                                float* __restrict__ agg,
                                                const float* __restrict__ dinv2,
                                                const float* __restrict__ b,
                                                int logF, size_t total) {
  int F1 = (1 << logF) - 1;
  for (size_t i = (size_t)blockIdx.x * blockDim.x + threadIdx.x; i < total;
       i += (size_t)gridDim.x * blockDim.x) {
    size_t n = i >> logF;
    int f = (int)(i & F1);
    agg[i] = h[i] * dinv2[n] + b[f];
  }
}

// agg[dst, :] += h[src, :] * norm[e], float4-vectorized, feature-coalesced
__global__ __launch_bounds__(256) void edge_scatter(const float4* __restrict__ h4,
                                                    float* __restrict__ agg,
                                                    const int* __restrict__ src,
                                                    const int* __restrict__ dst,
                                                    const float* __restrict__ nrm,
                                                    int clog2, size_t total) {
  int chunks = 1 << clog2;
  for (size_t i = (size_t)blockIdx.x * blockDim.x + threadIdx.x; i < total;
       i += (size_t)gridDim.x * blockDim.x) {
    size_t e = i >> clog2;
    int c = (int)(i & (chunks - 1));
    int s = src[e], d = dst[e];
    float w = nrm[e];
    float4 v = h4[(size_t)s * chunks + c];
    float* o = agg + ((size_t)d << (clog2 + 2)) + (c << 2);
    atomicAdd(o + 0, v.x * w);
    atomicAdd(o + 1, v.y * w);
    atomicAdd(o + 2, v.z * w);
    atomicAdd(o + 3, v.w * w);
  }
}

__global__ __launch_bounds__(256) void act_to_bf16(const float* __restrict__ agg,
                                                   unsigned short* __restrict__ xb,
                                                   size_t total) {
  for (size_t i = (size_t)blockIdx.x * blockDim.x + threadIdx.x; i < total;
       i += (size_t)gridDim.x * blockDim.x)
    xb[i] = f2bf(fmaxf(agg[i], 0.0f));
}

__global__ __launch_bounds__(256) void act_to_f32(const float* __restrict__ agg,
                                                  float* __restrict__ o, size_t total) {
  for (size_t i = (size_t)blockIdx.x * blockDim.x + threadIdx.x; i < total;
       i += (size_t)gridDim.x * blockDim.x)
    o[i] = fmaxf(agg[i], 0.0f);
}

// out[n, 0..2] = relu5[n, :] @ Wp[64,3] + bp
__global__ __launch_bounds__(256) void final_proj(const float* __restrict__ relu5,
                                                  const float* __restrict__ Wp,
                                                  const float* __restrict__ bp,
                                                  float* __restrict__ out, int n) {
  __shared__ float sW[192];
  __shared__ float sb[3];
  if (threadIdx.x < 192) sW[threadIdx.x] = Wp[threadIdx.x];
  if (threadIdx.x < 3) sb[threadIdx.x] = bp[threadIdx.x];
  __syncthreads();
  int i = blockIdx.x * blockDim.x + threadIdx.x;
  if (i >= n) return;
  const float* r = relu5 + (size_t)i * 64;
  float a0 = sb[0], a1 = sb[1], a2 = sb[2];
#pragma unroll 8
  for (int f = 0; f < 64; ++f) {
    float v = r[f];
    a0 += v * sW[f * 3 + 0];
    a1 += v * sW[f * 3 + 1];
    a2 += v * sW[f * 3 + 2];
  }
  out[(size_t)i * 3 + 0] = a0;
  out[(size_t)i * 3 + 1] = a1;
  out[(size_t)i * 3 + 2] = a2;
}

// ---------------- host launch ----------------

static inline unsigned gs_blocks(size_t total) {
  size_t b = (total + 255) / 256;
  return (unsigned)(b > 1048576 ? 1048576 : b);
}

extern "C" void kernel_launch(void* const* d_in, const int* in_sizes, int n_in,
                              void* d_out, int out_size, void* d_ws, size_t ws_size,
                              hipStream_t stream) {
  (void)in_sizes; (void)n_in; (void)out_size; (void)ws_size;
  const float* x  = (const float*)d_in[0];
  const int*   ei = (const int*)d_in[1];
  const float* ea = (const float*)d_in[2];
  const float* W[5]; const float* B[5];
  for (int l = 0; l < 5; ++l) {
    W[l] = (const float*)d_in[3 + 2 * l];
    B[l] = (const float*)d_in[4 + 2 * l];
  }
  const float* Wp = (const float*)d_in[13];
  const float* bp = (const float*)d_in[14];
  float* out = (float*)d_out;
  const int* src = ei;
  const int* dst = ei + NE;

  const int Fin[5]  = {512, 2048, 1024, 512, 128};
  const int Fout[5] = {2048, 1024, 512, 128, 64};

  // workspace carve-out (256B aligned)
  char* ws = (char*)d_ws;
  size_t off = 0;
  auto carve = [&](size_t bytes) -> char* {
    char* p = ws + off;
    off += (bytes + 255) & ~(size_t)255;
    return p;
  };
  unsigned short* xb   = (unsigned short*)carve((size_t)NN * 2048 * 2);
  float* h     = (float*)carve((size_t)NN * 2048 * 4);
  float* agg   = (float*)carve((size_t)NN * 2048 * 4);
  float* relu5 = (float*)carve((size_t)NN * 64 * 4);
  float* nrm   = (float*)carve((size_t)NE * 4);
  float* deg   = (float*)carve((size_t)NN * 4);
  float* dinv  = (float*)carve((size_t)NN * 4);
  float* dinv2 = (float*)carve((size_t)NN * 4);
  unsigned short* Wt[5];
  for (int l = 0; l < 5; ++l)
    Wt[l] = (unsigned short*)carve((size_t)Fin[l] * Fout[l] * 2);

  // --- precompute ---
  size_t xtot = (size_t)NN * 512;
  cvt_f32_to_bf16<<<gs_blocks(xtot), 256, 0, stream>>>(x, xb, xtot);
  for (int l = 0; l < 5; ++l) {
    size_t wtot = (size_t)Fin[l] * Fout[l];
    wt_transpose<<<gs_blocks(wtot), 256, 0, stream>>>(W[l], Wt[l], Fin[l], Fout[l]);
  }
  fill_ones<<<(NN + 255) / 256, 256, 0, stream>>>(deg, NN);
  deg_scatter<<<(NE + 255) / 256, 256, 0, stream>>>(deg, dst, ea, NE);
  dinv_kernel<<<(NN + 255) / 256, 256, 0, stream>>>(deg, dinv, dinv2, NN);
  norm_kernel<<<(NE + 255) / 256, 256, 0, stream>>>(dinv, src, dst, ea, nrm, NE);

  // --- layers ---
  const int strips = NN / 16;             // 3125 (exact)
  const unsigned gy = (strips + 7) / 8;   // 391
  for (int l = 0; l < 5; ++l) {
    int K = Fin[l], N = Fout[l];
    dim3 grid(N / 64, gy);
    gemm_bf16_wmma<<<grid, 256, 0, stream>>>(xb, Wt[l], h, NN, N, K);

    int logF = 31 - __builtin_clz((unsigned)N);
    size_t total = (size_t)NN * N;
    init_agg<<<gs_blocks(total), 256, 0, stream>>>(h, agg, dinv2, B[l], logF, total);

    int clog2 = logF - 2;
    size_t stotal = (size_t)NE << clog2;
    edge_scatter<<<gs_blocks(stotal), 256, 0, stream>>>(
        (const float4*)h, agg, src, dst, nrm, clog2, stotal);

    if (l < 4)
      act_to_bf16<<<gs_blocks(total), 256, 0, stream>>>(agg, xb, total);
    else
      act_to_f32<<<gs_blocks(total), 256, 0, stream>>>(agg, relu5, total);
  }

  final_proj<<<(NN + 255) / 256, 256, 0, stream>>>(relu5, Wp, bp, out, NN);
}
